// RNN_39865886441726
// MI455X (gfx1250) — matrix operand order repaired and checked by reading, not compiled, not run on hardware
//
#include <hip/hip_runtime.h>
#include <hip/hip_bf16.h>
#include <math.h>
#include <stdint.h>

#define SEQ   1024
#define BATCH 64
#define INF_  512     // input features
#define HID   1024

typedef __attribute__((ext_vector_type(16))) __bf16   v16bf;
typedef __attribute__((ext_vector_type(8)))  __bf16   v8bf;
typedef __attribute__((ext_vector_type(8)))  float    v8f;
typedef __attribute__((ext_vector_type(4)))  float    v4f;
typedef __attribute__((ext_vector_type(4)))  uint32_t u32x4;
typedef __attribute__((ext_vector_type(8)))  uint32_t u32x8;

// ---------------------------------------------------------------------------
// Init: convert W_hh fp32 -> bf16 (done once, then L2-resident for 1024 steps)
// and zero the initial hidden state (bf16).
// ---------------------------------------------------------------------------
__global__ void rnn_init_kernel(const float* __restrict__ Whh,
                                __bf16* __restrict__ Whh_bf,
                                __bf16* __restrict__ h0_bf) {
    int idx = blockIdx.x * blockDim.x + threadIdx.x;
    if (idx < HID * HID)   Whh_bf[idx] = (__bf16)Whh[idx];
    if (idx < BATCH * HID) h0_bf[idx]  = (__bf16)0.0f;
}

// ---------------------------------------------------------------------------
// A-fragment loader: fp32 row -> bf16 WMMA A layout
// (lane holds row lane&15, K chunk {kb..kb+15}, kb=(lane<16?0:16))
// ---------------------------------------------------------------------------
__device__ __forceinline__ v16bf load_a_frag_f32(const float* pa) {
    v4f a0 = *(const v4f*)(pa + 0);
    v4f a1 = *(const v4f*)(pa + 4);
    v4f a2 = *(const v4f*)(pa + 8);
    v4f a3 = *(const v4f*)(pa + 12);
    v16bf r;
#pragma unroll
    for (int i = 0; i < 4; ++i) {
        r[i + 0]  = (__bf16)a0[i];
        r[i + 4]  = (__bf16)a1[i];
        r[i + 8]  = (__bf16)a2[i];
        r[i + 12] = (__bf16)a3[i];
    }
    return r;
}

// ---------------------------------------------------------------------------
// Phase 1: x_proj = X @ Wih^T + bih + bhh.
// M=65536, K=512, N=1024.  WG = 128 thr (4 waves), WG tile 128(M) x 64(N);
// wave w owns rows {16w..16w+15} and {64+16w..64+16w+15} (8 accumulators),
// so each LDS-staged B tile feeds 8 WMMAs per wave per K-step.
// ---------------------------------------------------------------------------
__global__ void __launch_bounds__(128)
rnn_xproj_kernel(const float* __restrict__ X,     // [SEQ*BATCH, INF_]
                 const float* __restrict__ Wih,   // [HID, INF_] (row-major)
                 const float* __restrict__ bih,   // [HID]
                 const float* __restrict__ bhh,   // [HID]
                 float* __restrict__ Out)         // [SEQ*BATCH, HID]
{
    __shared__ __align__(32) __bf16 sB[64 * 32];  // 64 cols x 32 K, bf16

    const int lane  = threadIdx.x & 31;
    const int wave  = threadIdx.x >> 5;
    const int mBase = blockIdx.x * 128;
    const int nBase = blockIdx.y * 64;

    const int kbA = (lane < 16) ? 0 : 16;
    const int kbB = (lane < 16) ? 0 : 16;
    const int mRow0 = mBase + wave * 16 + (lane & 15);
    const int mRow1 = mRow0 + 64;

    v8f acc[2][4] = {};

    for (int k = 0; k < INF_; k += 32) {
        // ---- cooperative stage of Wih tile (64 x 32) -> LDS as bf16 ----
        {
            const int row  = threadIdx.x >> 1;          // 0..63
            const int koff = (threadIdx.x & 1) * 16;    // 0 or 16
            const float* src = Wih + (size_t)(nBase + row) * INF_ + k + koff;
            __bf16* dst = &sB[row * 32 + koff];
            v4f f0 = *(const v4f*)(src + 0);
            v4f f1 = *(const v4f*)(src + 4);
            v4f f2 = *(const v4f*)(src + 8);
            v4f f3 = *(const v4f*)(src + 12);
#pragma unroll
            for (int i = 0; i < 4; ++i) {
                dst[i + 0]  = (__bf16)f0[i];
                dst[i + 4]  = (__bf16)f1[i];
                dst[i + 8]  = (__bf16)f2[i];
                dst[i + 12] = (__bf16)f3[i];
            }
        }
        __syncthreads();

        const float* pa0 = X + (size_t)mRow0 * INF_ + k + kbA;
        const float* pa1 = X + (size_t)mRow1 * INF_ + k + kbA;
        v16bf afrag0 = load_a_frag_f32(pa0);
        v16bf afrag1 = load_a_frag_f32(pa1);
        if (k + 32 < INF_) {
            __builtin_prefetch(pa0 + 32, 0, 3);
            __builtin_prefetch(pa1 + 32, 0, 3);
        }

#pragma unroll
        for (int t = 0; t < 4; ++t) {
            v16bf bfrag = *(const v16bf*)&sB[(t * 16 + (lane & 15)) * 32 + kbB];
            acc[0][t] = __builtin_amdgcn_wmma_f32_16x16x32_bf16(
                false, afrag0, false, bfrag, (short)0, acc[0][t], false, false);
            acc[1][t] = __builtin_amdgcn_wmma_f32_16x16x32_bf16(
                false, afrag1, false, bfrag, (short)0, acc[1][t], false, false);
        }
        __syncthreads();
    }

    // ---- epilogue: + (bih+bhh), store fp32 ----
    const int rbase = (lane < 16) ? 0 : 8;
#pragma unroll
    for (int half = 0; half < 2; ++half) {
#pragma unroll
        for (int t = 0; t < 4; ++t) {
            const int n = nBase + t * 16 + (lane & 15);
            const float bias = bih[n] + bhh[n];
#pragma unroll
            for (int r = 0; r < 8; ++r) {
                const int m = mBase + half * 64 + wave * 16 + rbase + r;
                Out[(size_t)m * HID + n] = acc[half][t][r] + bias;
            }
        }
    }
}

// ---------------------------------------------------------------------------
// Phase 2: one launch per timestep.  out = tanh(x_proj + h_in @ W_hh^T).
// Grid = 16 WGs (N tiles of 64), WG = 128 thr (4 waves).
// The WG's whole W_hh slice (64 cols x 1024 K bf16 = 128 KB, fits in the
// 320 KB WGP LDS) is DMA'd once per step by the Tensor Data Mover
// (tensor_load_to_lds, 2-D descriptor), then the K-loop is barrier-free:
// ds_load_b128 B fragments + WMMA only — minimal latency on the serial path.
// ---------------------------------------------------------------------------
__global__ void __launch_bounds__(128)
rnn_step_kernel(const __bf16* __restrict__ Whh_bf, // [HID, HID] bf16
                const __bf16* __restrict__ hin,    // [BATCH, HID] bf16
                __bf16* __restrict__ hout,         // [BATCH, HID] bf16
                float* __restrict__ OutS)          // [BATCH, HID] fp32 (in: x_proj, out: h)
{
    __shared__ __align__(128) __bf16 sW[64 * HID];   // 128 KB: full W slice

    const int lane  = threadIdx.x & 31;
    const int wave  = threadIdx.x >> 5;
    const int nBase = blockIdx.x * 64;

    if (wave == 0) {
        // ---- TDM: DMA 64 x 1024 bf16 tile (global -> LDS), one instruction.
        const uint64_t ga  = (uint64_t)(uintptr_t)(Whh_bf + (size_t)nBase * HID);
        const uint32_t lds = (uint32_t)(uintptr_t)(&sW[0]);

        // D# group 0: count=1 | lds_addr | global_addr | type=2
        u32x4 g0;
        g0[0] = 1u;                                   // count=1, gather off
        g0[1] = lds;                                  // lds_addr (bytes)
        g0[2] = (uint32_t)ga;                         // global_addr[31:0]
        g0[3] = (uint32_t)(ga >> 32) | (2u << 30);    // global_addr[56:32] | type=2

        // D# group 1: data_size=2B, tensor 1024x64, tile 1024x64, stride=HID
        u32x8 g1;
        g1[0] = 1u << 16;                 // workgroup_mask=0, data_size=1 (2B)
        g1[1] = (uint32_t)HID << 16;      // tensor_dim0[15:0]=1024 at bits[63:48]
        g1[2] = (uint32_t)64  << 16;      // tensor_dim0 hi=0 | tensor_dim1[15:0]=64
        g1[3] = (uint32_t)HID << 16;      // tensor_dim1 hi=0 | tile_dim0=1024
        g1[4] = 64u;                      // tile_dim1=64, tile_dim2=0
        g1[5] = (uint32_t)HID;            // tensor_dim0_stride=1024 (elements)
        g1[6] = 0u;
        g1[7] = 0u;

        u32x4 gz = {0u, 0u, 0u, 0u};      // groups 2/3: tile_dim3/4=0 => unused (2-D)

        asm volatile("tensor_load_to_lds %0, %1, %2, %3"
                     :: "s"(g0), "s"(g1), "s"(gz), "s"(gz)
                     : "memory");
        __builtin_amdgcn_s_wait_tensorcnt(0);
    }
    __syncthreads();

    const int kbA = (lane < 16) ? 0 : 16;
    const int kbB = (lane < 16) ? 0 : 16;
    const int mRow = wave * 16 + (lane & 15);

    v8f acc[4] = {};
    const __bf16* paBase = hin + (size_t)mRow * HID + kbA;

#pragma unroll 4
    for (int k = 0; k < HID; k += 32) {
        const __bf16* pa = paBase + k;
        v8bf lo = *(const v8bf*)(pa + 0);
        v8bf hi = *(const v8bf*)(pa + 8);
        v16bf afrag;
#pragma unroll
        for (int i = 0; i < 8; ++i) { afrag[i] = lo[i]; afrag[8 + i] = hi[i]; }

        if (k + 32 < HID) __builtin_prefetch(pa + 32, 0, 3);

#pragma unroll
        for (int t = 0; t < 4; ++t) {
            v16bf bfrag = *(const v16bf*)&sW[(size_t)(t * 16 + (lane & 15)) * HID + k + kbB];
            acc[t] = __builtin_amdgcn_wmma_f32_16x16x32_bf16(
                false, afrag, false, bfrag, (short)0, acc[t], false, false);
        }
    }

    // ---- epilogue: h = tanh(x_proj + acc); write fp32 out + bf16 carry ----
    const int rbase = (lane < 16) ? 0 : 8;
#pragma unroll
    for (int t = 0; t < 4; ++t) {
        const int n = nBase + t * 16 + (lane & 15);
#pragma unroll
        for (int r = 0; r < 8; ++r) {
            const int m = wave * 16 + rbase + r;
            const size_t off = (size_t)m * HID + n;
            const float v = tanhf(OutS[off] + acc[t][r]);
            OutS[off] = v;
            hout[off] = (__bf16)v;
        }
    }
}

// ---------------------------------------------------------------------------
extern "C" void kernel_launch(void* const* d_in, const int* in_sizes, int n_in,
                              void* d_out, int out_size, void* d_ws, size_t ws_size,
                              hipStream_t stream) {
    const float* X   = (const float*)d_in[0];  // [SEQ, BATCH, INF_]
    const float* Wih = (const float*)d_in[1];  // [HID, INF_]
    const float* Whh = (const float*)d_in[2];  // [HID, HID]
    const float* bih = (const float*)d_in[3];  // [HID]
    const float* bhh = (const float*)d_in[4];  // [HID]
    float* Out = (float*)d_out;                // [SEQ, BATCH, HID]

    // Workspace layout (bf16): W_hh (2 MB) | h buf0 (128 KB) | h buf1 (128 KB)
    __bf16* Whh_bf = (__bf16*)d_ws;
    __bf16* hbuf0  = Whh_bf + (size_t)HID * HID;
    __bf16* hbuf1  = hbuf0 + (size_t)BATCH * HID;

    rnn_init_kernel<<<(HID * HID) / 256, 256, 0, stream>>>(Whh, Whh_bf, hbuf0);

    dim3 g1((SEQ * BATCH) / 128, HID / 64);    // 512 x 16 WGs
    rnn_xproj_kernel<<<g1, 128, 0, stream>>>(X, Wih, bih, bhh, Out);

    for (int s = 0; s < SEQ; ++s) {
        const __bf16* hin  = (s & 1) ? hbuf1 : hbuf0;
        __bf16*       hout = (s & 1) ? hbuf0 : hbuf1;
        rnn_step_kernel<<<HID / 64, 128, 0, stream>>>(
            Whh_bf, hin, hout, Out + (size_t)s * BATCH * HID);
    }
}
